// ProbSparseAttention_42597485642366
// MI455X (gfx1250) — matrix operand order repaired and checked by reading, not compiled, not run on hardware
//
#include <hip/hip_runtime.h>
#include <hip/hip_bf16.h>

// ---------------------------------------------------------------------------
// ProbSparse attention for MI455X (gfx1250, wave32, WMMA bf16 16x16x32)
// Global->LDS staging is load-all-then-store-all so multiple global_load_b128
// stay in flight (partial loadcnt waits) instead of one serialized quad.
// ---------------------------------------------------------------------------

typedef __attribute__((ext_vector_type(16))) __bf16 v16bf;
typedef __attribute__((ext_vector_type(8)))  float  v8f;
typedef unsigned short bf16b;   // raw bf16 bits

#define DEV __device__ __forceinline__

DEV bf16b f2bf(float f) {                    // round-to-nearest-even
  unsigned u = __float_as_uint(f);
  u += 0x7fffu + ((u >> 16) & 1u);
  return (bf16b)(u >> 16);
}
// order-preserving float<->uint transform for atomic max on LDS
DEV unsigned fenc(float f) {
  unsigned u = __float_as_uint(f);
  return (u & 0x80000000u) ? ~u : (u | 0x80000000u);
}
DEV float fdec(unsigned k) {
  unsigned u = (k & 0x80000000u) ? (k & 0x7fffffffu) : ~k;
  return __uint_as_float(u);
}

union FragU { v16bf v; uint4 q[2]; };

// A fragment: 16x32 bf16 tile, row-major source (ld in elements).
// lane L (m=L&15, h=L>>4): VGPR0-3 = A[m][ko+h*8 .. +7], VGPR4-7 = A[m][ko+16+h*8 .. +7]
DEV v16bf afrag(const bf16b* sm, int ld, int mo, int ko, int lane) {
  int m = lane & 15, h = lane >> 4;
  FragU f;
  const bf16b* p = sm + (mo + m) * ld + ko;
  f.q[0] = *(const uint4*)(p + h * 8);
  f.q[1] = *(const uint4*)(p + 16 + h * 8);
  return f.v;
}
// B fragment from B^T row-major (N x K): lane L (n=L&15, h=L>>4) holds
// VGPR0-3 = Bt[no+n][ko+h*8 .. +7], VGPR4-7 = Bt[no+n][ko+16+h*8 .. +7]
DEV v16bf bfrag(const bf16b* sm, int ld, int no, int ko, int lane) {
  int n = lane & 15, h = lane >> 4;
  FragU f;
  const bf16b* p = sm + (no + n) * ld + ko + h * 8;
  f.q[0] = *(const uint4*)(p);
  f.q[1] = *(const uint4*)(p + 16);
  return f.v;
}
DEV v8f wmma_bf(v16bf a, v16bf b, v8f c) {
  return __builtin_amdgcn_wmma_f32_16x16x32_bf16(false, a, false, b,
                                                 (short)0, c, false, false);
}

// ---------------------------------------------------------------------------
// elementwise fp32 -> bf16
__global__ __launch_bounds__(256) void cvt_bf16(const float* __restrict__ s,
                                                bf16b* __restrict__ d, int n) {
  int i = blockIdx.x * 256 + threadIdx.x;
  int stride = gridDim.x * 256;
  for (; i < n; i += stride) d[i] = f2bf(s[i]);
}

// transpose+convert 512x512 weight: d[n*512+k] = bf16(s[k*512+n])
__global__ __launch_bounds__(256) void cvtT_bf16_512(const float* __restrict__ s,
                                                     bf16b* __restrict__ d) {
  int e = blockIdx.x * 256 + threadIdx.x;   // 0..262143
  int n = e >> 9, k = e & 511;
  d[e] = f2bf(s[k * 512 + n]);
}

// ---------------------------------------------------------------------------
// Tiled bf16 WMMA GEMM: C(MxN) = A(MxK) * B(KxN) + bias, with Bt = B^T (NxK)
// Block: 256 thr / 8 waves, 128x128 tile, K-step 64.
__global__ __launch_bounds__(256) void gemm_bf16(
    const bf16b* __restrict__ A, const bf16b* __restrict__ Bt,
    const float* __restrict__ bias, float* __restrict__ C,
    int M, int N, int K) {
  __shared__ bf16b As[128 * 64];
  __shared__ bf16b Bs[128 * 64];
  const int t = threadIdx.x, lane = t & 31, w = t >> 5;
  const int wm = w & 3, wn = w >> 2;
  const int m0 = blockIdx.y * 128, n0 = blockIdx.x * 128;
  const bf16b* Abase = A + (size_t)m0 * K;
  const bf16b* Bbase = Bt + (size_t)n0 * K;

  v8f acc[2][4];
#pragma unroll
  for (int i = 0; i < 2; ++i)
#pragma unroll
    for (int j = 0; j < 4; ++j)
#pragma unroll
      for (int e = 0; e < 8; ++e) acc[i][j][e] = 0.f;

  for (int k0 = 0; k0 < K; k0 += 64) {
    __syncthreads();
    {   // batched copy: all loads in flight, then drain to LDS
      uint4 ta[4], tb[4];
#pragma unroll
      for (int jj = 0; jj < 4; ++jj) {
        int i = t + 256 * jj, row = i >> 3, seg = i & 7;
        ta[jj] = *(const uint4*)(Abase + (size_t)row * K + k0 + seg * 8);
        tb[jj] = *(const uint4*)(Bbase + (size_t)row * K + k0 + seg * 8);
      }
#pragma unroll
      for (int jj = 0; jj < 4; ++jj) {
        int i = t + 256 * jj, row = i >> 3, seg = i & 7;
        *(uint4*)(As + row * 64 + seg * 8) = ta[jj];
        *(uint4*)(Bs + row * 64 + seg * 8) = tb[jj];
      }
    }
    if (k0 + 64 < K) {   // warm L2 for the next K-tile (global_prefetch_b8)
      int row = t >> 1, seg = t & 1;
      __builtin_prefetch(Abase + (size_t)row * K + k0 + 64 + seg * 32, 0, 1);
      __builtin_prefetch(Bbase + (size_t)row * K + k0 + 64 + seg * 32, 0, 1);
    }
    __syncthreads();
#pragma unroll
    for (int ks = 0; ks < 64; ks += 32) {
      v16bf a0 = afrag(As, 64, 32 * wm, ks, lane);
      v16bf a1 = afrag(As, 64, 32 * wm + 16, ks, lane);
      v16bf b0 = bfrag(Bs, 64, 64 * wn, ks, lane);
      v16bf b1 = bfrag(Bs, 64, 64 * wn + 16, ks, lane);
      v16bf b2 = bfrag(Bs, 64, 64 * wn + 32, ks, lane);
      v16bf b3 = bfrag(Bs, 64, 64 * wn + 48, ks, lane);
      acc[0][0] = wmma_bf(a0, b0, acc[0][0]);
      acc[0][1] = wmma_bf(a0, b1, acc[0][1]);
      acc[0][2] = wmma_bf(a0, b2, acc[0][2]);
      acc[0][3] = wmma_bf(a0, b3, acc[0][3]);
      acc[1][0] = wmma_bf(a1, b0, acc[1][0]);
      acc[1][1] = wmma_bf(a1, b1, acc[1][1]);
      acc[1][2] = wmma_bf(a1, b2, acc[1][2]);
      acc[1][3] = wmma_bf(a1, b3, acc[1][3]);
    }
  }
  const int ln = lane & 15, h = lane >> 4;
#pragma unroll
  for (int i = 0; i < 2; ++i)
#pragma unroll
    for (int j = 0; j < 4; ++j) {
      int rowb = m0 + 32 * wm + 16 * i + 8 * h;
      int col = n0 + 64 * wn + 16 * j + ln;
      float bv = bias ? bias[col] : 0.f;
#pragma unroll
      for (int r = 0; r < 8; ++r)
        C[(size_t)(rowb + r) * N + col] = acc[i][j][r] + bv;
    }
}

// ---------------------------------------------------------------------------
// Per-row (dk=64) prep: 1/|q| norms, bf16 Q/K, normalized-K bf16, V^T bf16.
__global__ __launch_bounds__(256) void rows_prep(
    const float* __restrict__ Qf, const float* __restrict__ Kf,
    const float* __restrict__ Vf, bf16b* __restrict__ Qb,
    bf16b* __restrict__ Kb, bf16b* __restrict__ Knb,
    bf16b* __restrict__ Vtb, float* __restrict__ rq_inv) {
  int row = blockIdx.x * 256 + threadIdx.x;      // 0..131071 = bh*2048 + t
  if (row >= 64 * 2048) return;
  int bh = row >> 11, tt = row & 2047;
  size_t base = (size_t)row * 64;
  {  // Q
    float4 v[16];
    float sq = 0.f;
#pragma unroll
    for (int i = 0; i < 16; ++i) {
      v[i] = *(const float4*)(Qf + base + 4 * i);
      sq += v[i].x * v[i].x + v[i].y * v[i].y + v[i].z * v[i].z + v[i].w * v[i].w;
    }
    rq_inv[row] = rsqrtf(fmaxf(sq, 1e-12f));
#pragma unroll
    for (int i = 0; i < 16; ++i) {
      union { bf16b h[4]; uint2 u; } o;
      o.h[0] = f2bf(v[i].x); o.h[1] = f2bf(v[i].y);
      o.h[2] = f2bf(v[i].z); o.h[3] = f2bf(v[i].w);
      *(uint2*)(Qb + base + 4 * i) = o.u;
    }
  }
  {  // K: raw bf16 + normalized bf16
    float4 v[16];
    float sq = 0.f;
#pragma unroll
    for (int i = 0; i < 16; ++i) {
      v[i] = *(const float4*)(Kf + base + 4 * i);
      sq += v[i].x * v[i].x + v[i].y * v[i].y + v[i].z * v[i].z + v[i].w * v[i].w;
    }
    float kin = rsqrtf(fmaxf(sq, 1e-12f));
#pragma unroll
    for (int i = 0; i < 16; ++i) {
      union { bf16b h[4]; uint2 u; } o, on;
      o.h[0] = f2bf(v[i].x); o.h[1] = f2bf(v[i].y);
      o.h[2] = f2bf(v[i].z); o.h[3] = f2bf(v[i].w);
      on.h[0] = f2bf(v[i].x * kin); on.h[1] = f2bf(v[i].y * kin);
      on.h[2] = f2bf(v[i].z * kin); on.h[3] = f2bf(v[i].w * kin);
      *(uint2*)(Kb + base + 4 * i) = o.u;
      *(uint2*)(Knb + base + 4 * i) = on.u;
    }
  }
  {  // V transposed per bh: Vtb[bh][d][t]
    size_t tb = (size_t)bh * 131072 + tt;
#pragma unroll
    for (int i = 0; i < 16; ++i) {
      float4 v = *(const float4*)(Vf + base + 4 * i);
      Vtb[tb + (size_t)(4 * i + 0) * 2048] = f2bf(v.x);
      Vtb[tb + (size_t)(4 * i + 1) * 2048] = f2bf(v.y);
      Vtb[tb + (size_t)(4 * i + 2) * 2048] = f2bf(v.z);
      Vtb[tb + (size_t)(4 * i + 3) * 2048] = f2bf(v.w);
    }
  }
}

// ---------------------------------------------------------------------------
// max_scores[bh][q] = rq_inv * max_k (Q[q] . Khat[k])  via WMMA 128x128 tiles
__global__ __launch_bounds__(256) void score_max(
    const bf16b* __restrict__ Qb, const bf16b* __restrict__ Knb,
    const float* __restrict__ rq_inv, float* __restrict__ msc) {
  __shared__ bf16b Qs[128 * 64];
  __shared__ bf16b Ks[128 * 64];
  __shared__ unsigned smax[128];
  const int t = threadIdx.x, lane = t & 31, w = t >> 5;
  const int wm = w & 3, wn = w >> 2;
  const int bh = blockIdx.y, q0 = blockIdx.x * 128;
  const size_t qbase = (size_t)bh * 131072 + (size_t)q0 * 64;
  const bf16b* Kbase = Knb + (size_t)bh * 131072;
#pragma unroll
  for (int jj = 0; jj < 4; ++jj) {
    int i = t + 256 * jj;
    *(uint4*)(Qs + i * 8) = *(const uint4*)(Qb + qbase + (size_t)i * 8);
  }
  if (t < 128) smax[t] = 0u;
  const int ln = lane & 15, h = lane >> 4;

  for (int kt = 0; kt < 16; ++kt) {
    __syncthreads();
    {
      uint4 tk[4];
#pragma unroll
      for (int jj = 0; jj < 4; ++jj) {
        int i = t + 256 * jj;
        tk[jj] = *(const uint4*)(Kbase + (size_t)kt * 8192 + (size_t)i * 8);
      }
#pragma unroll
      for (int jj = 0; jj < 4; ++jj) {
        int i = t + 256 * jj;
        *(uint4*)(Ks + i * 8) = tk[jj];
      }
    }
    if (kt < 15)   // prefetch next key stripe
      __builtin_prefetch(Kbase + (size_t)(kt + 1) * 8192 + (size_t)t * 32, 0, 1);
    __syncthreads();
    v8f acc[2][4];
#pragma unroll
    for (int i = 0; i < 2; ++i)
#pragma unroll
      for (int j = 0; j < 4; ++j)
#pragma unroll
        for (int e = 0; e < 8; ++e) acc[i][j][e] = 0.f;
#pragma unroll
    for (int ks = 0; ks < 64; ks += 32) {
      v16bf a0 = afrag(Qs, 64, 32 * wm, ks, lane);
      v16bf a1 = afrag(Qs, 64, 32 * wm + 16, ks, lane);
      v16bf b0 = bfrag(Ks, 64, 64 * wn, ks, lane);
      v16bf b1 = bfrag(Ks, 64, 64 * wn + 16, ks, lane);
      v16bf b2 = bfrag(Ks, 64, 64 * wn + 32, ks, lane);
      v16bf b3 = bfrag(Ks, 64, 64 * wn + 48, ks, lane);
      acc[0][0] = wmma_bf(a0, b0, acc[0][0]);
      acc[0][1] = wmma_bf(a0, b1, acc[0][1]);
      acc[0][2] = wmma_bf(a0, b2, acc[0][2]);
      acc[0][3] = wmma_bf(a0, b3, acc[0][3]);
      acc[1][0] = wmma_bf(a1, b0, acc[1][0]);
      acc[1][1] = wmma_bf(a1, b1, acc[1][1]);
      acc[1][2] = wmma_bf(a1, b2, acc[1][2]);
      acc[1][3] = wmma_bf(a1, b3, acc[1][3]);
    }
    // reduce max over keys: fold n-tiles, shuffle over 16 lanes, LDS atomic
#pragma unroll
    for (int i = 0; i < 2; ++i) {
      float m8[8];
#pragma unroll
      for (int r = 0; r < 8; ++r)
        m8[r] = fmaxf(fmaxf(acc[i][0][r], acc[i][1][r]),
                      fmaxf(acc[i][2][r], acc[i][3][r]));
#pragma unroll
      for (int off = 8; off >= 1; off >>= 1)
#pragma unroll
        for (int r = 0; r < 8; ++r)
          m8[r] = fmaxf(m8[r], __shfl_xor(m8[r], off, 16));
      if (ln == 0) {
        int rowb = 32 * wm + 16 * i + 8 * h;
#pragma unroll
        for (int r = 0; r < 8; ++r) atomicMax(&smax[rowb + r], fenc(m8[r]));
      }
    }
  }
  __syncthreads();
  if (t < 128) {
    int q = q0 + t;
    msc[bh * 2048 + q] = fdec(smax[t]) * rq_inv[bh * 2048 + q];
  }
}

// ---------------------------------------------------------------------------
// top-64 per bh (iterative argmax, lowest-index ties like jax top_k set)
__global__ __launch_bounds__(256) void topk64(const float* __restrict__ msc,
                                              int* __restrict__ tix) {
  __shared__ float vals[2048];
  __shared__ float rv[256];
  __shared__ int ri[256];
  const int t = threadIdx.x, bh = blockIdx.x;
#pragma unroll
  for (int i = 0; i < 8; ++i)
    vals[t + 256 * i] = msc[bh * 2048 + t + 256 * i];
  __syncthreads();
  for (int it = 0; it < 64; ++it) {
    float bv = -3.4e38f;
    int bi = 0x7fffffff;
#pragma unroll
    for (int i = 0; i < 8; ++i) {
      int idx = t + 256 * i;
      float v = vals[idx];
      if (v > bv) { bv = v; bi = idx; }
    }
    rv[t] = bv; ri[t] = bi;
    __syncthreads();
    for (int s = 128; s > 0; s >>= 1) {
      if (t < s) {
        float ov = rv[t + s]; int oi = ri[t + s];
        if (ov > rv[t] || (ov == rv[t] && oi < ri[t])) { rv[t] = ov; ri[t] = oi; }
      }
      __syncthreads();
    }
    if (t == 0) { tix[bh * 64 + it] = ri[0]; vals[ri[0]] = -3.4e38f; }
    __syncthreads();
  }
}

// ---------------------------------------------------------------------------
// S[bh] = Qsel(64x64) . K^T(64x2048) / 8, plus per-row max
__global__ __launch_bounds__(256) void sattn_scores(
    const float* __restrict__ Qf, const bf16b* __restrict__ Kb,
    const int* __restrict__ tix, float* __restrict__ Sb,
    float* __restrict__ rmx) {
  __shared__ bf16b Qs[64 * 64];
  __shared__ bf16b Ks[128 * 64];
  __shared__ unsigned smax[64];
  const int t = threadIdx.x, lane = t & 31, w = t >> 5;
  const int wm = w & 3, wn = w >> 2;
  const int bh = blockIdx.x;
  const bf16b* Kbase = Kb + (size_t)bh * 131072;
  float* S = Sb + (size_t)bh * 64 * 2048;
#pragma unroll
  for (int i = 0; i < 16; ++i) {   // gather selected queries, cvt to bf16
    int e = t + 256 * i, r = e >> 6, c = e & 63;
    int q = tix[bh * 64 + r];
    Qs[e] = f2bf(Qf[(size_t)bh * 131072 + (size_t)q * 64 + c]);
  }
  if (t < 64) smax[t] = 0u;
  const int ln = lane & 15, h = lane >> 4;

  for (int kt = 0; kt < 16; ++kt) {
    __syncthreads();
    {
      uint4 tk[4];
#pragma unroll
      for (int jj = 0; jj < 4; ++jj) {
        int i = t + 256 * jj;
        tk[jj] = *(const uint4*)(Kbase + (size_t)kt * 8192 + (size_t)i * 8);
      }
#pragma unroll
      for (int jj = 0; jj < 4; ++jj) {
        int i = t + 256 * jj;
        *(uint4*)(Ks + i * 8) = tk[jj];
      }
    }
    if (kt < 15)
      __builtin_prefetch(Kbase + (size_t)(kt + 1) * 8192 + (size_t)t * 32, 0, 1);
    __syncthreads();
    v8f acc[4];
#pragma unroll
    for (int j = 0; j < 4; ++j)
#pragma unroll
      for (int e = 0; e < 8; ++e) acc[j][e] = 0.f;
#pragma unroll
    for (int ks = 0; ks < 64; ks += 32) {
      v16bf a = afrag(Qs, 64, 16 * wm, ks, lane);
#pragma unroll
      for (int j = 0; j < 4; ++j) {
        v16bf b = bfrag(Ks, 64, 16 * (4 * wn + j), ks, lane);
        acc[j] = wmma_bf(a, b, acc[j]);
      }
    }
    float m8[8];
#pragma unroll
    for (int r = 0; r < 8; ++r) {
      float mv = -3.4e38f;
      int row = 16 * wm + 8 * h + r;
#pragma unroll
      for (int j = 0; j < 4; ++j) {
        float v = acc[j][r] * 0.125f;   // 1/sqrt(dk)
        S[(size_t)row * 2048 + kt * 128 + 16 * (4 * wn + j) + ln] = v;
        mv = fmaxf(mv, v);
      }
      m8[r] = mv;
    }
#pragma unroll
    for (int off = 8; off >= 1; off >>= 1)
#pragma unroll
      for (int r = 0; r < 8; ++r)
        m8[r] = fmaxf(m8[r], __shfl_xor(m8[r], off, 16));
    if (ln == 0) {
#pragma unroll
      for (int r = 0; r < 8; ++r)
        atomicMax(&smax[16 * wm + 8 * h + r], fenc(m8[r]));
    }
  }
  __syncthreads();
  if (t < 64) rmx[bh * 64 + t] = fdec(smax[t]);
}

// ---------------------------------------------------------------------------
// softmax row (2048) -> bf16 P
__global__ __launch_bounds__(256) void softmax_p(const float* __restrict__ Sb,
                                                 const float* __restrict__ rmx,
                                                 bf16b* __restrict__ Pb) {
  __shared__ float red[256];
  const int t = threadIdx.x, rowg = blockIdx.x;   // 0..4095 = bh*64 + row
  const float m = rmx[rowg];
  const size_t base = (size_t)rowg * 2048;
  float e[8], s = 0.f;
#pragma unroll
  for (int i = 0; i < 8; ++i) {
    e[i] = __expf(Sb[base + t + 256 * i] - m);
    s += e[i];
  }
  red[t] = s;
  __syncthreads();
  for (int sd = 128; sd > 0; sd >>= 1) {
    if (t < sd) red[t] += red[t + sd];
    __syncthreads();
  }
  float inv = 1.f / red[0];
#pragma unroll
  for (int i = 0; i < 8; ++i) Pb[base + t + 256 * i] = f2bf(e[i] * inv);
}

// ---------------------------------------------------------------------------
// O[bh] = P(64x2048) . V(2048x64), scattered into out_full bf16 rows
__global__ __launch_bounds__(256) void pv_scatter(
    const bf16b* __restrict__ Pb, const bf16b* __restrict__ Vtb,
    const int* __restrict__ tix, bf16b* __restrict__ outb) {
  __shared__ bf16b Ps[64 * 64];
  __shared__ bf16b Vs[64 * 64];
  const int t = threadIdx.x, lane = t & 31, w = t >> 5;
  const int wm = w & 3, wn = w >> 2;
  const int bh = blockIdx.x;
  const bf16b* Pbase = Pb + (size_t)bh * 64 * 2048;
  const bf16b* Vbase = Vtb + (size_t)bh * 131072;
  v8f acc[2];
#pragma unroll
  for (int j = 0; j < 2; ++j)
#pragma unroll
    for (int e = 0; e < 8; ++e) acc[j][e] = 0.f;

  for (int k0 = 0; k0 < 2048; k0 += 64) {
    __syncthreads();
    {
      uint4 tp[2], tv[2];
#pragma unroll
      for (int jj = 0; jj < 2; ++jj) {
        int i = t + 256 * jj, row = i >> 3, seg = i & 7;
        tp[jj] = *(const uint4*)(Pbase + (size_t)row * 2048 + k0 + seg * 8);
        tv[jj] = *(const uint4*)(Vbase + (size_t)row * 2048 + k0 + seg * 8);
      }
#pragma unroll
      for (int jj = 0; jj < 2; ++jj) {
        int i = t + 256 * jj, row = i >> 3, seg = i & 7;
        *(uint4*)(Ps + row * 64 + seg * 8) = tp[jj];
        *(uint4*)(Vs + row * 64 + seg * 8) = tv[jj];
      }
    }
    if (k0 + 64 < 2048) {
      int row = t >> 2, seg = t & 3;   // 64 rows x 4 x 16-elem chunks
      __builtin_prefetch(Pbase + (size_t)row * 2048 + k0 + 64 + seg * 16, 0, 1);
      __builtin_prefetch(Vbase + (size_t)row * 2048 + k0 + 64 + seg * 16, 0, 1);
    }
    __syncthreads();
#pragma unroll
    for (int ks = 0; ks < 64; ks += 32) {
      v16bf a = afrag(Ps, 64, 16 * wm, ks, lane);
#pragma unroll
      for (int j = 0; j < 2; ++j) {
        v16bf b = bfrag(Vs, 64, 16 * (2 * wn + j), ks, lane);
        acc[j] = wmma_bf(a, b, acc[j]);
      }
    }
  }
  const int ln = lane & 15, h = lane >> 4;
#pragma unroll
  for (int j = 0; j < 2; ++j)
#pragma unroll
    for (int r = 0; r < 8; ++r) {
      int orow = 16 * wm + 8 * h + r;
      int col = 16 * (2 * wn + j) + ln;
      int trow = tix[bh * 64 + orow];
      outb[(size_t)bh * 131072 + (size_t)trow * 64 + col] = f2bf(acc[j][r]);
    }
}

// ---------------------------------------------------------------------------
extern "C" void kernel_launch(void* const* d_in, const int* in_sizes, int n_in,
                              void* d_out, int out_size, void* d_ws,
                              size_t ws_size, hipStream_t stream) {
  (void)in_sizes; (void)n_in; (void)out_size; (void)ws_size;
  const float* x  = (const float*)d_in[0];
  const float* Wq = (const float*)d_in[1];
  const float* bq = (const float*)d_in[2];
  const float* Wk = (const float*)d_in[3];
  const float* bk = (const float*)d_in[4];
  const float* Wv = (const float*)d_in[5];
  const float* bv = (const float*)d_in[6];
  const float* Wo = (const float*)d_in[7];
  const float* bo = (const float*)d_in[8];
  float* out = (float*)d_out;

  const int M = 16384, N = 512, K = 512;   // B*T, D, D
  const size_t NE = (size_t)M * N;         // 8,388,608

  char* p = (char*)d_ws;
  auto alloc = [&](size_t bytes) -> void* {
    void* r = (void*)p;
    p += (bytes + 255) & ~(size_t)255;
    return r;
  };
  bf16b* xb   = (bf16b*)alloc(NE * 2);
  bf16b* Wqt  = (bf16b*)alloc((size_t)512 * 512 * 2);
  bf16b* Wkt  = (bf16b*)alloc((size_t)512 * 512 * 2);
  bf16b* Wvt  = (bf16b*)alloc((size_t)512 * 512 * 2);
  bf16b* Wot  = (bf16b*)alloc((size_t)512 * 512 * 2);
  float* Qf   = (float*)alloc(NE * 4);
  float* Kf   = (float*)alloc(NE * 4);
  float* Vf   = (float*)alloc(NE * 4);
  bf16b* Qb   = (bf16b*)alloc(NE * 2);
  bf16b* Kb   = (bf16b*)alloc(NE * 2);
  bf16b* Knb  = (bf16b*)alloc(NE * 2);
  bf16b* Vtb  = (bf16b*)alloc(NE * 2);
  float* rq   = (float*)alloc((size_t)64 * 2048 * 4);
  float* msc  = (float*)alloc((size_t)64 * 2048 * 4);
  int*   tix  = (int*)alloc((size_t)64 * 64 * 4);
  float* rmx  = (float*)alloc((size_t)64 * 64 * 4);
  float* Sb   = (float*)alloc((size_t)64 * 64 * 2048 * 4);
  bf16b* Pb   = (bf16b*)alloc((size_t)64 * 64 * 2048 * 2);
  bf16b* outb = (bf16b*)alloc(NE * 2);

  cvt_bf16<<<4096, 256, 0, stream>>>(x, xb, (int)NE);
  cvtT_bf16_512<<<1024, 256, 0, stream>>>(Wq, Wqt);
  cvtT_bf16_512<<<1024, 256, 0, stream>>>(Wk, Wkt);
  cvtT_bf16_512<<<1024, 256, 0, stream>>>(Wv, Wvt);
  cvtT_bf16_512<<<1024, 256, 0, stream>>>(Wo, Wot);

  dim3 gg(N / 128, M / 128);
  gemm_bf16<<<gg, 256, 0, stream>>>(xb, Wqt, bq, Qf, M, N, K);
  gemm_bf16<<<gg, 256, 0, stream>>>(xb, Wkt, bk, Kf, M, N, K);
  gemm_bf16<<<gg, 256, 0, stream>>>(xb, Wvt, bv, Vf, M, N, K);

  rows_prep<<<512, 256, 0, stream>>>(Qf, Kf, Vf, Qb, Kb, Knb, Vtb, rq);

  dim3 sg(16, 64);
  score_max<<<sg, 256, 0, stream>>>(Qb, Knb, rq, msc);
  topk64<<<64, 256, 0, stream>>>(msc, tix);
  sattn_scores<<<64, 256, 0, stream>>>(Qf, Kb, tix, Sb, rmx);
  softmax_p<<<4096, 256, 0, stream>>>(Sb, rmx, Pb);

  (void)hipMemsetAsync(outb, 0, NE * 2, stream);
  pv_scatter<<<64, 256, 0, stream>>>(Pb, Vtb, tix, outb);

  gemm_bf16<<<gg, 256, 0, stream>>>(outb, Wot, bo, out, M, N, K);
}
